// SineKANLayer_19258633355997
// MI455X (gfx1250) — compile-verified, matrix operand order checked by reading, not run on hardware
//
#include <hip/hip_runtime.h>
#include <cmath>

typedef __attribute__((ext_vector_type(16))) _Float16 v16h;
typedef __attribute__((ext_vector_type(8)))  _Float16 v8h;
typedef __attribute__((ext_vector_type(4)))  _Float16 v4h;
typedef __attribute__((ext_vector_type(8)))  float    v8f;
typedef __attribute__((ext_vector_type(4)))  float    v4f;
typedef __attribute__((ext_vector_type(2)))  float    v2f;

#define B_DIM   8192
#define IN_DIM  1024
#define OUT_DIM 512
#define GRID_N  8
#define K_DIM   (IN_DIM * GRID_N)   // 8192
#define BM      128
#define BN      128
#define KC      32                  // one WMMA K-step (f16)

__launch_bounds__(256)
__global__ void sinekan_fused_wmma(const float* __restrict__ x,
                                   const float* __restrict__ amp,
                                   const float* __restrict__ freq,
                                   const float* __restrict__ bias,
                                   float* __restrict__ out,
                                   float scale, float pj_coef)
{
    // A tile: 128 rows x 32 K halves, row-major (row stride KC halves = 64B)
    __shared__ _Float16 lds_a[BM * KC] __attribute__((aligned(64)));
    // B tile: 128 cols x 32 K halves, [n][k] (col stride KC halves = 64B)
    __shared__ _Float16 lds_b[BN * KC] __attribute__((aligned(64)));

    const int tid  = threadIdx.x;
    const int lane = tid & 31;
    const int wave = tid >> 5;

    const int bm0 = blockIdx.x * BM;   // batch-row base
    const int o0  = blockIdx.y * BN;   // output-col base

    // 8 waves: 4 along M (32 rows each) x 2 along N (64 cols each)
    const int wm = wave & 3;
    const int wn = wave >> 2;

    // sine-basis frequencies (8 floats, input array)
    float fr[GRID_N];
#pragma unroll
    for (int d = 0; d < GRID_N; ++d) fr[d] = freq[d];

    // 2x4 array of 16x16 f32 accumulators
    v8f acc[2][4];
    const v8f vzero = {0.f, 0.f, 0.f, 0.f, 0.f, 0.f, 0.f, 0.f};
#pragma unroll
    for (int tm = 0; tm < 2; ++tm)
#pragma unroll
        for (int tn = 0; tn < 4; ++tn) acc[tm][tn] = vzero;

    // A-fill mapping: thread -> (row, which pair of j's)
    const int am  = tid & (BM - 1);   // 0..127
    const int ajh = tid >> 7;         // 0..1 : j-offset pair {2*ajh, 2*ajh+1}

    // B-fill mapping: 8 threads cover the 32 K-values of one column
    const int bkk   = (tid & 7) * 4;  // K sub-offset (4 floats)
    const int bcbase = tid >> 3;      // column base 0..31 (+32 per iter)

    // WMMA fragment addressing (contiguous k = 16*(lane/16) + s, same for A and B)
    const int frow = lane & 15;            // M row / N col within tile
    const int fkh  = (lane >> 4) * 16;     // K-half select (halves)

    for (int k0 = 0; k0 < K_DIM; k0 += KC) {
        const int j0 = k0 >> 3;            // k0 / GRID_N : 4 j's per K-step

        // ---- A tile: on-the-fly sine basis, f32 -> f16 ----
        {
            const int j = j0 + ajh * 2;
            v2f xv = *(const v2f*)(x + (size_t)(bm0 + am) * IN_DIM + j);
            v16h av;
#pragma unroll
            for (int jj = 0; jj < 2; ++jj) {
                const float t  = xv[jj] + scale;              // x + ratio^(G-1)
                const float pj = pj_coef * (float)(j + jj);   // scale*pi*j/1023
#pragma unroll
                for (int d = 0; d < GRID_N; ++d)
                    av[jj * 8 + d] = (_Float16)__sinf(fr[d] * t + pj);
            }
            // 32-byte store: row am, halves [ajh*16, ajh*16+16)
            *(v16h*)(&lds_a[am * KC + ajh * 16]) = av;
        }

        // ---- B tile: amplitudes (O,K) row-major -> [n][k] f16 in LDS ----
#pragma unroll
        for (int ci = 0; ci < 4; ++ci) {
            const int c = bcbase + ci * 32;                   // 0..127
            const float* gsrc = amp + (size_t)(o0 + c) * K_DIM + k0 + bkk;
            v4f bv = *(const v4f*)gsrc;
            // prefetch next K-chunk of this column
            __builtin_prefetch(gsrc + KC, 0, 3);
            v4h hv;
#pragma unroll
            for (int i = 0; i < 4; ++i) hv[i] = (_Float16)bv[i];
            *(v4h*)(&lds_b[c * KC + bkk]) = hv;
        }

        __syncthreads();

        // ---- fragment loads + 2x4 WMMA ----
        v16h afrag[2], bfrag[4];
#pragma unroll
        for (int tm = 0; tm < 2; ++tm)
            afrag[tm] = *(const v16h*)(&lds_a[(wm * 32 + tm * 16 + frow) * KC + fkh]);
#pragma unroll
        for (int tn = 0; tn < 4; ++tn)
            bfrag[tn] = *(const v16h*)(&lds_b[(wn * 64 + tn * 16 + frow) * KC + fkh]);

#pragma unroll
        for (int tm = 0; tm < 2; ++tm)
#pragma unroll
            for (int tn = 0; tn < 4; ++tn)
                acc[tm][tn] = __builtin_amdgcn_wmma_f32_16x16x32_f16(
                    /*neg_a=*/false, afrag[tm],
                    /*neg_b=*/false, bfrag[tn],
                    /*c_mod=*/(short)0, acc[tm][tn],
                    /*reuse_a=*/false, /*reuse_b=*/false);

        __syncthreads();
    }

    // ---- epilogue: add bias, store f32 ----
    // C/D layout: VGPR r -> row (r + 8*(lane/16)), col = lane%16
    const int ccol  = lane & 15;
    const int rbase = (lane >> 4) * 8;
#pragma unroll
    for (int tn = 0; tn < 4; ++tn) {
        const int o  = o0 + wn * 64 + tn * 16 + ccol;
        const float bv = bias[o];
#pragma unroll
        for (int tm = 0; tm < 2; ++tm) {
            const int mrow = bm0 + wm * 32 + tm * 16 + rbase;
#pragma unroll
            for (int r = 0; r < 8; ++r)
                out[(size_t)(mrow + r) * OUT_DIM + o] = acc[tm][tn][r] + bv;
        }
    }
}

extern "C" void kernel_launch(void* const* d_in, const int* in_sizes, int n_in,
                              void* d_out, int out_size, void* d_ws, size_t ws_size,
                              hipStream_t stream)
{
    (void)in_sizes; (void)n_in; (void)d_ws; (void)ws_size; (void)out_size;

    const float* x    = (const float*)d_in[0];   // (8192, 1024)
    const float* amp  = (const float*)d_in[1];   // (512, 1024, 8) = (O, K)
    const float* freq = (const float*)d_in[2];   // (8,)
    const float* bias = (const float*)d_in[3];   // (512,)
    float* out = (float*)d_out;                  // (8192, 512)

    // ratio = A_C * G^(-K_C) + C_C ; scale = ratio^(G-1) ; computed in double
    const double ratio = 0.9724 * pow((double)GRID_N, -0.9884) + 0.9994;
    const float scale   = (float)pow(ratio, (double)(GRID_N - 1));
    const float pj_coef = (float)(pow(ratio, (double)(GRID_N - 1)) *
                                  M_PI / (double)(IN_DIM - 1));

    dim3 grid(B_DIM / BM, OUT_DIM / BN);   // 64 x 4
    dim3 block(256);                        // 8 wave32
    sinekan_fused_wmma<<<grid, block, 0, stream>>>(x, amp, freq, bias, out,
                                                   scale, pj_coef);
}
